// EvenOddEdgeUpdateLayer_28149215658675
// MI455X (gfx1250) — matrix operand order repaired and checked by reading, not compiled, not run on hardware
//
#include <hip/hip_runtime.h>
#include <hip/hip_bf16.h>

// ---------------------------------------------------------------------------
// EvenOddEdgeUpdateLayer on MI455X (gfx1250), bf16 WMMA path.
//   out = nbr_fea + ( softplus( [nbr|even_i|even_j|odd_i*odd_j] @ W1 + b1 ) @ W2 + b2 )
// One wave32 = one node (its 16 neighbor edges = one 16-row WMMA A tile).
// A fragments are cached in registers across the 4 output-column tiles.
// ---------------------------------------------------------------------------

typedef __attribute__((ext_vector_type(16))) __bf16 bf16x16;
typedef __attribute__((ext_vector_type(8)))  float  f32x8;

#define WAVES  8      // waves per block (256 threads)
#define XSTR   264    // X tile row stride in bf16 elems (16B-aligned rows, padded)
#define W1STR  264    // W1^T row stride
#define HSTR   72     // H tile row stride
#define W2STR  72     // W2^T row stride

// Native bf16 truncation (RNE fptrunc; lowers to v_cvt_*bf16_f32 when available)
__device__ __forceinline__ unsigned short f2bf(float f) {
  union { __bf16 h; unsigned short u; } r;
  r.h = (__bf16)f;
  return r.u;
}

// Pack two f32 -> one dword of 2x bf16
__device__ __forceinline__ unsigned pk2bf(float x, float y) {
  union { __bf16 h[2]; unsigned u; } r;
  r.h[0] = (__bf16)x;
  r.h[1] = (__bf16)y;
  return r.u;
}

union FragU { uint4 q[2]; bf16x16 v; };

// A layout (16-bit A 16x32, wave32): lane l (h=l>>4, r=l&15) holds
//   elems 0..7  = A[r][32*kb + 8h + 0..7]
//   elems 8..15 = A[r][32*kb + 16 + 8h + 0..7]
__device__ __forceinline__ bf16x16 load_a_frag(const unsigned short* base, int row,
                                               int stride, int kb, int h) {
  FragU f;
  const unsigned short* p = base + row * stride + kb * 32 + h * 8;
  f.q[0] = *(const uint4*)p;
  f.q[1] = *(const uint4*)(p + 16);
  return f.v;
}

// B layout (16-bit B 32x16, wave32): lane l (h=l>>4, n=l&15) holds
//   elems 0..7  = B[32*kb + 8h + 0..7][n]
//   elems 8..15 = B[32*kb + 16 + 8h + 0..7][n]
__device__ __forceinline__ bf16x16 load_b_frag(const unsigned short* base, int col,
                                               int stride, int kb, int h) {
  FragU f;
  const unsigned short* p = base + col * stride + kb * 32 + h * 8;
  f.q[0] = *(const uint4*)p;
  f.q[1] = *(const uint4*)(p + 16);
  return f.v;
}

__device__ __forceinline__ void cvt32_store(const float* __restrict__ src,
                                            unsigned short* dst) {
  #pragma unroll
  for (int q = 0; q < 8; ++q) {
    float4 f = ((const float4*)src)[q];
    ((uint2*)dst)[q] = make_uint2(pk2bf(f.x, f.y), pk2bf(f.z, f.w));
  }
}

__device__ __forceinline__ void cvt32_mul_store(const float* __restrict__ s0,
                                                const float* __restrict__ s1,
                                                unsigned short* dst) {
  #pragma unroll
  for (int q = 0; q < 8; ++q) {
    float4 a = ((const float4*)s0)[q];
    float4 b = ((const float4*)s1)[q];
    ((uint2*)dst)[q] = make_uint2(pk2bf(a.x * b.x, a.y * b.y),
                                  pk2bf(a.z * b.z, a.w * b.w));
  }
}

// fast softplus: max(x,0) + log(1 + exp(-|x|)) using HW transcendentals
// (TRANS32 ops co-execute with WMMA on CDNA5; error << bf16 GEMM quantization)
__device__ __forceinline__ float softplus_fast(float x) {
  return fmaxf(x, 0.f) + __logf(1.0f + __expf(-fabsf(x)));
}

__global__ void __launch_bounds__(256, 1)
eoe_edge_update_kernel(const float* __restrict__ even_node,
                       const float* __restrict__ odd_node,
                       const float* __restrict__ nbr_fea,
                       const int*   __restrict__ nbr_idx,
                       const float* __restrict__ W1, const float* __restrict__ b1,
                       const float* __restrict__ W2, const float* __restrict__ b2,
                       float* __restrict__ out, int N) {
  __shared__ __align__(16) unsigned short w1t[64 * W1STR];          // 33 KB
  __shared__ __align__(16) unsigned short w2t[64 * W2STR];          //  9 KB
  __shared__ __align__(16) unsigned short xbuf[WAVES * 16 * XSTR];  // 66 KB
  __shared__ __align__(16) unsigned short hbuf[WAVES * 16 * HSTR];  // 18 KB

  const int tid = threadIdx.x;

  // ---- stage W1^T, W2^T (bf16) once per block ----
  for (int idx = tid; idx < 256 * 64; idx += 256) {
    int k = idx >> 6, n = idx & 63;
    w1t[n * W1STR + k] = f2bf(W1[idx]);
  }
  for (int idx = tid; idx < 64 * 64; idx += 256) {
    int k = idx >> 6, n = idx & 63;
    w2t[n * W2STR + k] = f2bf(W2[idx]);
  }
  __syncthreads();

  const int wave = tid >> 5;
  const int lane = tid & 31;
  const int lrow = lane & 15;     // A row / B column within 16-wide tile
  const int h    = lane >> 4;     // half-wave select

  unsigned short* xw = xbuf + wave * 16 * XSTR;
  unsigned short* hw = hbuf + wave * 16 * HSTR;

  // loop-invariant per-lane biases (hoisted out of the node loop)
  float b1v[4], b2v[4];
  #pragma unroll
  for (int nt = 0; nt < 4; ++nt) {
    b1v[nt] = b1[nt * 16 + lrow];
    b2v[nt] = b2[nt * 16 + lrow];
  }

  const int stride = gridDim.x * WAVES;
  for (int node = blockIdx.x * WAVES + wave; node < N; node += stride) {
    // prefetch next node's edge features into cache hierarchy
    if (node + stride < N)
      __builtin_prefetch(nbr_fea + (size_t)(node + stride) * 16 * 64, 0, 3);

    // ---- stage X tile: 16 rows x 256 bf16 (8 chunks of 32 per row) ----
    #pragma unroll
    for (int job = lane; job < 128; job += 32) {
      int m = job >> 3;           // neighbor slot / tile row
      int c = job & 7;            // 32-col chunk
      unsigned short* dst = xw + m * XSTR + c * 32;
      if (c < 2) {
        cvt32_store(nbr_fea + ((size_t)(node * 16 + m)) * 64 + c * 32, dst);
      } else if (c < 4) {
        cvt32_store(even_node + (size_t)node * 64 + (c - 2) * 32, dst);
      } else if (c < 6) {
        int j = nbr_idx[node * 16 + m];
        cvt32_store(even_node + (size_t)j * 64 + (c - 4) * 32, dst);
      } else {
        int j = nbr_idx[node * 16 + m];
        cvt32_mul_store(odd_node + (size_t)node * 64 + (c - 6) * 32,
                        odd_node + (size_t)j    * 64 + (c - 6) * 32, dst);
      }
    }
    asm volatile("s_wait_dscnt 0" ::: "memory");   // wave-local LDS RAW fence

    // ---- GEMM1: H = softplus(X @ W1 + b1), K=256 ----
    // Cache the 8 A fragments once; reuse across all 4 output-column tiles.
    bf16x16 a1[8];
    #pragma unroll
    for (int kb = 0; kb < 8; ++kb)
      a1[kb] = load_a_frag(xw, lrow, XSTR, kb, h);

    #pragma unroll
    for (int nt = 0; nt < 4; ++nt) {
      f32x8 acc = {};
      #pragma unroll
      for (int kb = 0; kb < 8; ++kb) {
        bf16x16 b = load_b_frag(w1t, nt * 16 + lrow, W1STR, kb, h);
        acc = __builtin_amdgcn_wmma_f32_16x16x32_bf16(
                  false, a1[kb], false, b, (short)0, acc, false, false);
      }
      #pragma unroll
      for (int v = 0; v < 8; ++v) {
        float sp = softplus_fast(acc[v] + b1v[nt]);
        // C/D layout: VGPR v, lane l -> element (M = v + 8h, N = nt*16 + (l&15))
        hw[(v + 8 * h) * HSTR + nt * 16 + lrow] = f2bf(sp);
      }
    }
    asm volatile("s_wait_dscnt 0" ::: "memory");

    // ---- GEMM2: delta = H @ W2 + b2 (K=64), residual add in fp32 ----
    bf16x16 a2[2];
    #pragma unroll
    for (int kb = 0; kb < 2; ++kb)
      a2[kb] = load_a_frag(hw, lrow, HSTR, kb, h);

    #pragma unroll
    for (int nt = 0; nt < 4; ++nt) {
      f32x8 acc = {};
      #pragma unroll
      for (int kb = 0; kb < 2; ++kb) {
        bf16x16 b = load_b_frag(w2t, nt * 16 + lrow, W2STR, kb, h);
        acc = __builtin_amdgcn_wmma_f32_16x16x32_bf16(
                  false, a2[kb], false, b, (short)0, acc, false, false);
      }
      #pragma unroll
      for (int v = 0; v < 8; ++v) {
        int m = v + 8 * h;
        size_t off = ((size_t)(node * 16 + m)) * 64 + nt * 16 + lrow;
        out[off] = nbr_fea[off] + acc[v] + b2v[nt];   // exact fp32 residual
      }
    }
  }
}

extern "C" void kernel_launch(void* const* d_in, const int* in_sizes, int n_in,
                              void* d_out, int out_size, void* d_ws, size_t ws_size,
                              hipStream_t stream) {
  const float* even_node = (const float*)d_in[0];
  const float* odd_node  = (const float*)d_in[1];
  const float* nbr_fea   = (const float*)d_in[2];
  const int*   nbr_idx   = (const int*)  d_in[3];
  const float* W1        = (const float*)d_in[4];
  const float* b1        = (const float*)d_in[5];
  const float* W2        = (const float*)d_in[6];
  const float* b2        = (const float*)d_in[7];
  float*       out       = (float*)d_out;

  int N = in_sizes[0] / 64;                  // nodes
  int blocks = (N + WAVES - 1) / WAVES;
  if (blocks > 1024) blocks = 1024;          // persistent blocks; amortize W staging
  if (blocks < 1) blocks = 1;

  eoe_edge_update_kernel<<<blocks, 256, 0, stream>>>(
      even_node, odd_node, nbr_fea, nbr_idx, W1, b1, W2, b2, out, N);
}
